// ImprovedSlidingWindowGQA_80315888435254
// MI455X (gfx1250) — matrix operand order repaired and checked by reading, not compiled, not run on hardware
//
#include <hip/hip_runtime.h>
#include <hip/hip_bf16.h>
#include <math.h>

typedef __attribute__((ext_vector_type(16))) _Float16 v16h;
typedef __attribute__((ext_vector_type(8)))  float    v8f;
typedef unsigned int u32;
typedef __attribute__((ext_vector_type(4))) u32 v4u;
typedef __attribute__((ext_vector_type(8))) int v8i;
typedef __attribute__((ext_vector_type(4))) int v4i;

#define DIMSZ   1024
#define NH      16
#define NKV     4
#define HD      64
#define LSEQ    2048
#define BATCH   2
#define NTOT    1536      /* fused q|k|v columns: 1024+256+256 */
#define HALFWIN 256
#define NGLOB   64
#define EPSV    1e-6f

static __device__ inline v8f wmma_f16(v16h a, v16h b, v8f c) {
  return __builtin_amdgcn_wmma_f32_16x16x32_f16(false, a, false, b, (short)0, c,
                                                false, false);
}

// ---------------------------------------------------------------------------
// Tensor Data Mover: issue a 2D f16 tile load (tile_d1 rows x tile_d0 cols)
// from a row-major tensor (row length stride0 elements) into LDS at lds_off.
// D# bit layout per CDNA5 ISA 8.3/8.4. 2D tensor -> groups 2/3 zero.
// ---------------------------------------------------------------------------
static __device__ inline void tdm_load_2d(u32 lds_off, const void* gaddr,
                                          u32 tensor_d0, u32 tensor_d1,
                                          u32 tile_d0, u32 tile_d1, u32 stride0) {
  const unsigned long long ga = (unsigned long long)gaddr;
  v4u g0;
  g0[0] = 1u;                               // count=1, user descriptor
  g0[1] = lds_off;                          // LDS byte address
  g0[2] = (u32)ga;                          // global_addr[31:0]
  g0[3] = (u32)(ga >> 32) | (2u << 30);     // global_addr[56:32] | type=2
  v8i g1;
  g1[0] = (int)(1u << 16);                  // wg_mask=0, data_size=1 (2 bytes)
  g1[1] = (int)((tensor_d0 & 0xffffu) << 16);            // tensor_dim0[15:0]
  g1[2] = (int)((tensor_d0 >> 16) | ((tensor_d1 & 0xffffu) << 16));
  g1[3] = (int)((tensor_d1 >> 16) | (tile_d0 << 16));    // tile_dim0
  g1[4] = (int)tile_d1;                     // tile_dim1 (tile_dim2=0 -> 2D)
  g1[5] = (int)stride0;                     // tensor_dim0_stride[31:0]
  g1[6] = 0;
  g1[7] = 0;
  v4i z = {0, 0, 0, 0};
#if __clang_major__ >= 23
  v8i z8 = {0, 0, 0, 0, 0, 0, 0, 0};
  __builtin_amdgcn_tensor_load_to_lds(g0, g1, z, z, z8, 0);
#else
  __builtin_amdgcn_tensor_load_to_lds(g0, g1, z, z, 0);
#endif
}

#define LDS_OFF(p) ((u32)(unsigned long long)(const void*)(p))

// ---------------------------------------------------------------------------
// f16 GEMM: C(MxN,f32) = A(MxK,f16) * B(KxN,f16). 256 threads / 8 waves,
// 64x64 tile, K-step 32. Tiles staged to LDS by the Tensor Data Mover
// (double buffered, wave 0 drives the DMA, everyone syncs on TENSORcnt).
// ---------------------------------------------------------------------------
#define BM 64
#define BN 64
#define BK 32

__global__ __launch_bounds__(256)
void gemm_wmma_h(const _Float16* __restrict__ A, const _Float16* __restrict__ B,
                 float* __restrict__ C, int M, int N, int K) {
  __shared__ _Float16 As[2][BM * BK];
  __shared__ _Float16 Bs[2][BK * BN];
  const int tid  = threadIdx.x;
  const int lane = tid & 31;
  const int wave = tid >> 5;
  const int lm   = lane & 15;
  const int half = lane >> 4;
  const int m0 = blockIdx.y * BM;
  const int n0 = blockIdx.x * BN;
  const int wm = (wave >> 2) * 32;   // 0 or 32
  const int wn = (wave & 3) * 16;    // 0,16,32,48
  v8f acc0 = {};
  v8f acc1 = {};
  const int nsteps = K / BK;

  if (wave == 0) {  // prologue: stage 0 (TDM ignores EXEC; 1 instr = 1 DMA)
    tdm_load_2d(LDS_OFF(&As[0][0]), A + (size_t)m0 * K, (u32)K, (u32)M,
                BK, BM, (u32)K);
    tdm_load_2d(LDS_OFF(&Bs[0][0]), B + n0, (u32)N, (u32)K,
                BN, BK, (u32)N);
  }
  for (int s = 0; s < nsteps; ++s) {
    const int cur = s & 1;
    if (wave == 0) {
      if (s + 1 < nsteps) {
        const int k1 = (s + 1) * BK;
        tdm_load_2d(LDS_OFF(&As[1 - cur][0]), A + (size_t)m0 * K + k1,
                    (u32)K, (u32)M, BK, BM, (u32)K);
        tdm_load_2d(LDS_OFF(&Bs[1 - cur][0]), B + (size_t)k1 * N + n0,
                    (u32)N, (u32)K, BN, BK, (u32)N);
        __builtin_amdgcn_s_wait_tensorcnt((short)2);  // stage s landed
      } else {
        __builtin_amdgcn_s_wait_tensorcnt((short)0);
      }
    }
    __syncthreads();
    const int kb = half * 16;
    v16h a0, a1, bf;
#pragma unroll
    for (int e = 0; e < 8; ++e) {
      a0[e]     = As[cur][(wm + lm) * BK + kb + e];
      a0[e + 8] = As[cur][(wm + lm) * BK + kb + 8 + e];
      a1[e]     = As[cur][(wm + 16 + lm) * BK + kb + e];
      a1[e + 8] = As[cur][(wm + 16 + lm) * BK + kb + 8 + e];
    }
    const int koff = half * 16;
#pragma unroll
    for (int e = 0; e < 16; ++e) bf[e] = Bs[cur][(koff + e) * BN + wn + lm];
    acc0 = wmma_f16(a0, bf, acc0);
    acc1 = wmma_f16(a1, bf, acc1);
    __syncthreads();
  }
  const int crow = half * 8;
#pragma unroll
  for (int i = 0; i < 8; ++i) {
    C[(size_t)(m0 + wm + crow + i) * N + n0 + wn + lm]      = acc0[i];
    C[(size_t)(m0 + wm + 16 + crow + i) * N + n0 + wn + lm] = acc1[i];
  }
}

// ---------------------------------------------------------------------------
// f32 -> f16 matrix convert with column remap (builds xh / [wq|wk|wv] / woh).
// ---------------------------------------------------------------------------
__global__ void convert_mat(const float* __restrict__ src, _Float16* __restrict__ dst,
                            int ncols, int dstride, int doff, int total) {
  const int idx = blockIdx.x * blockDim.x + threadIdx.x;
  if (idx >= total) return;
  const int r = idx / ncols;
  const int c = idx - r * ncols;
  dst[(size_t)r * dstride + doff + c] = (_Float16)src[idx];
}

// ---------------------------------------------------------------------------
// RMSNorm + RoPE from fused qkv f32 (row stride NTOT) -> f16 (B, Hc, L, 64).
// One wave per (token, head); lane t owns RoPE pair (2t, 2t+1).
// ---------------------------------------------------------------------------
__global__ __launch_bounds__(32)
void normrope_kernel(const float* __restrict__ src, int rstride, int coff,
                     const float* __restrict__ g, _Float16* __restrict__ dst,
                     int Hc) {
  const int bid = blockIdx.x;
  const int t   = threadIdx.x;        // 0..31
  const int h   = bid % Hc;
  const int tok = bid / Hc;
  const int b   = tok / LSEQ;
  const int l   = tok - b * LSEQ;
  const float* sp = src + (size_t)tok * rstride + coff + h * HD;
  const float x1 = sp[2 * t];
  const float x2 = sp[2 * t + 1];
  float ss = x1 * x1 + x2 * x2;
#pragma unroll
  for (int m = 16; m >= 1; m >>= 1) ss += __shfl_xor(ss, m, 32);
  const float r = rsqrtf(ss * (1.0f / HD) + EPSV);
  const float y1 = x1 * r * g[2 * t];
  const float y2 = x2 * r * g[2 * t + 1];
  const float inv = __expf(-0.28782313662425f * (float)t);  // 10000^(-t/32)
  const float ang = (float)l * inv;
  float sn, cs;
  __sincosf(ang, &sn, &cs);
  _Float16* op = dst + (((size_t)b * Hc + h) * LSEQ + l) * HD;
  op[2 * t]     = (_Float16)(y1 * cs - y2 * sn);
  op[2 * t + 1] = (_Float16)(y1 * sn + y2 * cs);
}

// v slice of fused qkv f32 -> vh f16 (B, NKV, L, 64)
__global__ void convert_v_kernel(const float* __restrict__ qkv,
                                 _Float16* __restrict__ vh) {
  const int idx = blockIdx.x * blockDim.x + threadIdx.x;
  const int total = BATCH * LSEQ * NKV * HD;
  if (idx >= total) return;
  const int d    = idx & (HD - 1);
  const int rest = idx >> 6;
  const int hv   = rest % NKV;
  const int tok  = rest / NKV;
  const int b    = tok / LSEQ;
  const int l    = tok - b * LSEQ;
  vh[(((size_t)b * NKV + hv) * LSEQ + l) * HD + d] =
      (_Float16)qkv[(size_t)tok * NTOT + DIMSZ + NKV * HD + hv * HD + d];
}

// ---------------------------------------------------------------------------
// Windowed GQA attention. One wave per (b, h, 16-query block).
// Scores via 2x WMMA (K=64), masked f32 scores in LDS, softmax, f16 P in LDS,
// P*V via 4x WMMA per 32-key step. Output f16 (B, L, H*64).
// ---------------------------------------------------------------------------
#define QB 16
#define MAXNK 352   // 22 blocks * 16, multiple of 32

__global__ __launch_bounds__(32)
void attn_kernel(const _Float16* __restrict__ qh, const _Float16* __restrict__ kh,
                 const _Float16* __restrict__ vh, _Float16* __restrict__ attnh) {
  __shared__ float    S[QB][MAXNK];
  __shared__ _Float16 P[QB][MAXNK];
  const int nqb = LSEQ / QB;                 // 128
  const int qb  = blockIdx.x % nqb;
  const int bh  = blockIdx.x / nqb;
  const int b   = bh / NH;
  const int h   = bh % NH;
  const int kvh = h >> 2;                    // REP = 4
  const int lane = threadIdx.x;
  const int lm   = lane & 15;
  const int half = lane >> 4;
  const int qstart = qb * QB;

  const _Float16* qp = qh + (((size_t)b * NH + h) * LSEQ + qstart) * HD;
  const _Float16* kp = kh + (((size_t)b * NKV + kvh) * LSEQ) * HD;
  const _Float16* vp = vh + (((size_t)b * NKV + kvh) * LSEQ) * HD;

  const int kb = half * 16;
  v16h aq0, aq1;
#pragma unroll
  for (int e = 0; e < 8; ++e) {
    aq0[e]     = qp[lm * HD + kb + e];
    aq0[e + 8] = qp[lm * HD + kb + 8 + e];
    aq1[e]     = qp[lm * HD + 32 + kb + e];
    aq1[e + 8] = qp[lm * HD + 32 + kb + 8 + e];
  }

  int kb_lo = qstart - HALFWIN; if (kb_lo < 0) kb_lo = 0; kb_lo >>= 4;
  const int kb_hi = (qstart + QB - 1) >> 4;
  int ng = kb_lo; if (ng > NGLOB / 16) ng = NGLOB / 16;
  const int nb = ng + (kb_hi - kb_lo + 1);
  const int nk = nb * 16;

  const int koff = half * 16;
  for (int bi = 0; bi < nb; ++bi) {
    const int key0 = ((bi < ng) ? bi : (kb_lo + bi - ng)) * 16;
    v16h bk0, bk1;
    const _Float16* krow = kp + (size_t)(key0 + lm) * HD;
#pragma unroll
    for (int e = 0; e < 16; ++e) {
      bk0[e] = krow[koff + e];
      bk1[e] = krow[32 + koff + e];
    }
    v8f sc = {};
    sc = wmma_f16(aq0, bk0, sc);
    sc = wmma_f16(aq1, bk1, sc);
    const int col = bi * 16 + lm;
#pragma unroll
    for (int i = 0; i < 8; ++i) {
      const int iq = qstart + half * 8 + i;
      const int j  = key0 + lm;
      const bool ok = (j <= iq) && ((j >= iq - HALFWIN) || (j < NGLOB));
      S[half * 8 + i][col] = ok ? sc[i] * 0.125f : -1e30f;
    }
  }
  __syncthreads();

  const int r = lm;   // lanes (r, r+16) cooperate on row r
  float mx = -1e30f;
  for (int c = half; c < nk; c += 2) mx = fmaxf(mx, S[r][c]);
  mx = fmaxf(mx, __shfl_xor(mx, 16, 32));
  float sum = 0.0f;
  for (int c = half; c < nk; c += 2) sum += __expf(S[r][c] - mx);
  sum += __shfl_xor(sum, 16, 32);
  const float inv_l = 1.0f / sum;
  for (int c = half; c < nk; c += 2)
    P[r][c] = (_Float16)(__expf(S[r][c] - mx) * inv_l);
  const int nkpad = (nk + 31) & ~31;
  for (int c = nk + half; c < nkpad; c += 2) P[r][c] = (_Float16)0.0f;
  __syncthreads();

  v8f o0 = {}, o1 = {}, o2 = {}, o3 = {};
  for (int kk = 0; kk < nkpad; kk += 32) {
    v16h ap;
#pragma unroll
    for (int e = 0; e < 8; ++e) {
      ap[e]     = P[lm][kk + kb + e];
      ap[e + 8] = P[lm][kk + kb + 8 + e];
    }
    v16h bv0, bv1, bv2, bv3;
#pragma unroll
    for (int e = 0; e < 16; ++e) {
      const int c   = kk + koff + e;
      const int bi2 = c >> 4;
      const int k0  = ((bi2 < ng) ? bi2 : (kb_lo + bi2 - ng)) * 16;
      int j = k0 + (c & 15);
      if (j >= LSEQ) j = LSEQ - 1;    // pad region: P==0, keep address in range
      const _Float16* vr = vp + (size_t)j * HD;
      bv0[e] = vr[lm];
      bv1[e] = vr[16 + lm];
      bv2[e] = vr[32 + lm];
      bv3[e] = vr[48 + lm];
    }
    o0 = wmma_f16(ap, bv0, o0);
    o1 = wmma_f16(ap, bv1, o1);
    o2 = wmma_f16(ap, bv2, o2);
    o3 = wmma_f16(ap, bv3, o3);
  }

  _Float16* op = attnh + ((size_t)b * LSEQ + qstart) * DIMSZ + h * HD;
#pragma unroll
  for (int i = 0; i < 8; ++i) {
    const int row = half * 8 + i;
    op[(size_t)row * DIMSZ + lm]      = (_Float16)o0[i];
    op[(size_t)row * DIMSZ + 16 + lm] = (_Float16)o1[i];
    op[(size_t)row * DIMSZ + 32 + lm] = (_Float16)o2[i];
    op[(size_t)row * DIMSZ + 48 + lm] = (_Float16)o3[i];
  }
}

// ---------------------------------------------------------------------------
extern "C" void kernel_launch(void* const* d_in, const int* in_sizes, int n_in,
                              void* d_out, int out_size, void* d_ws, size_t ws_size,
                              hipStream_t stream) {
  (void)in_sizes; (void)n_in; (void)out_size; (void)ws_size;
  const float* x  = (const float*)d_in[0];
  const float* wq = (const float*)d_in[1];
  const float* wk = (const float*)d_in[2];
  const float* wv = (const float*)d_in[3];
  const float* wo = (const float*)d_in[4];
  const float* gq = (const float*)d_in[5];
  const float* gk = (const float*)d_in[6];
  float* out = (float*)d_out;

  const int M = BATCH * LSEQ;                         // 4096 tokens
  const size_t kvsz = (size_t)M * (NKV * HD);         // 1M elems
  float* qkv = (float*)d_ws;                          // M*1536 f32
  _Float16* xh    = (_Float16*)(qkv + (size_t)M * NTOT);   // M*1024 f16
  _Float16* wqkvh = xh + (size_t)M * DIMSZ;                // 1024*1536 f16
  _Float16* woh   = wqkvh + (size_t)DIMSZ * NTOT;          // 1024*1024 f16
  _Float16* qh    = woh + (size_t)DIMSZ * DIMSZ;           // M*1024 f16
  _Float16* kh    = qh + (size_t)M * DIMSZ;                // M*256 f16
  _Float16* vh    = kh + kvsz;                             // M*256 f16
  _Float16* attnh = vh + kvsz;                             // M*1024 f16

  // one-time f16 packing (x once, weights once)
  convert_mat<<<(M * DIMSZ + 255) / 256, 256, 0, stream>>>(x, xh, DIMSZ, DIMSZ, 0, M * DIMSZ);
  convert_mat<<<(DIMSZ * DIMSZ + 255) / 256, 256, 0, stream>>>(wq, wqkvh, DIMSZ, NTOT, 0, DIMSZ * DIMSZ);
  convert_mat<<<(DIMSZ * NKV * HD + 255) / 256, 256, 0, stream>>>(wk, wqkvh, NKV * HD, NTOT, DIMSZ, DIMSZ * NKV * HD);
  convert_mat<<<(DIMSZ * NKV * HD + 255) / 256, 256, 0, stream>>>(wv, wqkvh, NKV * HD, NTOT, DIMSZ + NKV * HD, DIMSZ * NKV * HD);
  convert_mat<<<(DIMSZ * DIMSZ + 255) / 256, 256, 0, stream>>>(wo, woh, DIMSZ, DIMSZ, 0, DIMSZ * DIMSZ);

  // fused QKV projection: reads x once (f16), TDM-staged WMMA GEMM
  gemm_wmma_h<<<dim3(NTOT / BN, M / BM), 256, 0, stream>>>(xh, wqkvh, qkv, M, NTOT, DIMSZ);

  // RMSNorm + RoPE (+ transpose to head-major), V convert
  normrope_kernel<<<M * NH, 32, 0, stream>>>(qkv, NTOT, 0, gq, qh, NH);
  normrope_kernel<<<M * NKV, 32, 0, stream>>>(qkv, NTOT, DIMSZ, gk, kh, NKV);
  convert_v_kernel<<<(M * NKV * HD + 255) / 256, 256, 0, stream>>>(qkv, vh);

  // windowed GQA attention (WMMA scores + WMMA PV), f16 out
  attn_kernel<<<BATCH * NH * (LSEQ / QB), 32, 0, stream>>>(qh, kh, vh, attnh);

  // output projection
  gemm_wmma_h<<<dim3(DIMSZ / BN, M / BM), 256, 0, stream>>>(attnh, woh, out, M, DIMSZ, DIMSZ);
}
